// Res3DGraphModel_62672162783765
// MI455X (gfx1250) — compile-verified, matrix operand order checked by reading, not run on hardware
//
#include <hip/hip_runtime.h>
#include <hip/hip_bf16.h>

// ---------------- problem constants (from the reference) ----------------
#define NNODES   10000
#define NEDGES   80000
#define NVOCAB   26
#define RES_EMB  64
#define EDGE_DIM 64
#define NHEADS   8
#define CUTOFF_F 6.0f
#define NEG_SLP  0.01f
#define MAXHC    2048      // HEADS * max C
#define MAXD     256       // max node feature width
#define NT       4         // N-tiles (16 cols each) per wave in the GEMM

typedef __bf16 v16bf __attribute__((ext_vector_type(16)));
typedef float  v8f   __attribute__((ext_vector_type(8)));

// ---------------- helpers ----------------
__device__ __forceinline__ unsigned flipf(float f) {
  unsigned u = __float_as_uint(f);
  return (u & 0x80000000u) ? ~u : (u | 0x80000000u);   // monotone float->uint
}
__device__ __forceinline__ float unflipf(unsigned u) {
  return __uint_as_float((u & 0x80000000u) ? (u & 0x7FFFFFFFu) : ~u);
}
__device__ __forceinline__ void atomicAddF32(float* p, float v) {
  __hip_atomic_fetch_add(p, v, __ATOMIC_RELAXED, __HIP_MEMORY_SCOPE_AGENT);
}

// ---------------- weight swizzle: f32 [K,N] -> bf16 WMMA-B fragment layout ----------
// out element index = ((nt*nKt + kt)*32 + lane)*16 + e
// lane: n = nt*16 + lane%16, h = lane/16
// e<8  -> k = kt*32 + 8h + e ; e>=8 -> k = kt*32 + 16 + 8h + (e-8)
__global__ void k_swizzle_w(const float* __restrict__ W, __hip_bfloat16* __restrict__ out,
                            int K, int Nc) {
  int idx = blockIdx.x * blockDim.x + threadIdx.x;
  if (idx >= K * Nc) return;
  int e    = idx & 15;
  int lane = (idx >> 4) & 31;
  int t    = idx >> 9;
  int nKt  = K >> 5;
  int kt   = t % nKt;
  int nt   = t / nKt;
  int h    = lane >> 4;
  int n    = nt * 16 + (lane & 15);
  int k    = kt * 32 + ((e < 8) ? (8 * h + e) : (16 + 8 * h + (e - 8)));
  out[idx] = __float2bfloat16(W[(size_t)k * Nc + n]);
}

// ---------------- activation swizzle: f32 [M,K] -> bf16 WMMA-A fragment layout ------
// out element index = ((mt*nKt + kt)*32 + lane)*16 + e
// lane: m = mt*16 + lane%16, h = lane/16; same K mapping as B.
__global__ void k_swizzle_a(const float* __restrict__ A, __hip_bfloat16* __restrict__ out,
                            int M, int K) {
  int idx = blockIdx.x * blockDim.x + threadIdx.x;
  if (idx >= M * K) return;
  int e    = idx & 15;
  int lane = (idx >> 4) & 31;
  int t    = idx >> 9;
  int nKt  = K >> 5;
  int kt   = t % nKt;
  int mt   = t / nKt;
  int h    = lane >> 4;
  int m    = mt * 16 + (lane & 15);
  int k    = kt * 32 + ((e < 8) ? (8 * h + e) : (16 + 8 * h + (e - 8)));
  out[idx] = __float2bfloat16(A[(size_t)m * K + k]);
}

// ---------------- WMMA GEMM: Y[M,Nc] = A @ B + bias, both pre-swizzled bf16 ---------
// One wave computes a 16 x (NT*16) output strip; K mult of 64 (nKt even), Nc mult of NT*16.
// Ping-pong software pipeline with two fixed fragment sets (no register rotation).
__global__ void k_wmma_gemm(const __hip_bfloat16* __restrict__ Aswz,
                            const __hip_bfloat16* __restrict__ Bswz,
                            const float* __restrict__ bias,
                            float* __restrict__ Y,
                            int M, int K, int Nc) {
  const int lane = threadIdx.x & 31;
  const int wave = blockIdx.x * (blockDim.x >> 5) + (threadIdx.x >> 5);
  const int nTn  = Nc >> 4;        // 16-col tiles
  const int nTg  = nTn >> 2;       // groups of NT tiles
  const int nGroups = (M >> 4) * nTg;
  if (wave >= nGroups) return;     // uniform per wave: EXEC stays all-1s for WMMA
  const int mt = wave / nTg;
  const int ng = wave - mt * nTg;
  const int nKt = K >> 5;          // even for all GEMMs in this model

  const __hip_bfloat16* __restrict__ ap =
      Aswz + (((size_t)mt * nKt) * 32 + lane) * 16;
  const __hip_bfloat16* __restrict__ bp =
      Bswz + ((((size_t)ng * NT) * nKt) * 32 + lane) * 16;
  const size_t bTileStride = (size_t)nKt * 512;   // elements between N-tiles

  v8f acc[NT];
#pragma unroll
  for (int i = 0; i < NT; ++i) acc[i] = (v8f){0.f,0.f,0.f,0.f,0.f,0.f,0.f,0.f};

  // prologue: fragment set A holds kt = 0
  v16bf aA = *(const v16bf*)ap;
  v16bf bA[NT];
#pragma unroll
  for (int i = 0; i < NT; ++i)
    bA[i] = *(const v16bf*)(bp + i * bTileStride);

  for (int kt = 0; kt < nKt; kt += 2) {
    // set B <- kt+1 (always valid: nKt even), in flight during set-A WMMAs
    const size_t o1 = (size_t)(kt + 1) * 512;
    const v16bf aB = *(const v16bf*)(ap + o1);
    v16bf bB[NT];
#pragma unroll
    for (int i = 0; i < NT; ++i)
      bB[i] = *(const v16bf*)(bp + i * bTileStride + o1);

#pragma unroll
    for (int i = 0; i < NT; ++i)
      acc[i] = __builtin_amdgcn_wmma_f32_16x16x32_bf16(false, aA, false, bA[i],
                                                       (short)0, acc[i], false, false);

    // set A <- kt+2, in flight during set-B WMMAs
    if (kt + 2 < nKt) {
      const size_t o2 = (size_t)(kt + 2) * 512;
      __builtin_prefetch(ap + o2 + 1024, 0, 0);     // global_prefetch_b8, two steps out
      aA = *(const v16bf*)(ap + o2);
#pragma unroll
      for (int i = 0; i < NT; ++i)
        bA[i] = *(const v16bf*)(bp + i * bTileStride + o2);
    }

#pragma unroll
    for (int i = 0; i < NT; ++i)
      acc[i] = __builtin_amdgcn_wmma_f32_16x16x32_bf16(false, aB, false, bB[i],
                                                       (short)0, acc[i], false, false);
  }

  // C/D layout: VGPR vg, lane -> M = vg + 8*(lane/16), N = lane%16
  const int r = lane & 15;
  const int h = lane >> 4;
#pragma unroll
  for (int i = 0; i < NT; ++i) {
    const int ncol = (ng * NT + i) * 16 + r;
    const float badd = bias ? bias[ncol] : 0.f;
#pragma unroll
    for (int vg = 0; vg < 8; ++vg) {
      const int orow = mt * 16 + vg + 8 * h;
      Y[(size_t)orow * Nc + ncol] = acc[i][vg] + badd;
    }
  }
}

// ---------------- node embedding: x = residue_atom @ node_W + node_b (K=26, tiny) ---
__global__ void k_node_embed(const float* __restrict__ ra, const float* __restrict__ W,
                             const float* __restrict__ b, float* __restrict__ x) {
  int idx = blockIdx.x * blockDim.x + threadIdx.x;
  if (idx >= NNODES * RES_EMB) return;
  int n = idx >> 6;
  int j = idx & 63;
  float acc = b[j];
#pragma unroll
  for (int v = 0; v < NVOCAB; ++v)
    acc += ra[n * NVOCAB + v] * W[v * RES_EMB + j];
  x[idx] = acc;
}

// ---------------- gaussian smearing: [E,64] ----------------
__global__ void k_smear(const float* __restrict__ d, float* __restrict__ ea) {
  int idx = blockIdx.x * blockDim.x + threadIdx.x;
  if (idx >= NEDGES * EDGE_DIM) return;
  int e = idx >> 6, g = idx & 63;
  const float step  = CUTOFF_F / (float)(EDGE_DIM - 1);
  const float coeff = -0.5f / (step * step);
  float t = d[e] - step * (float)g;
  ea[idx] = __expf(coeff * t * t);
}

// ---------------- attention logits + segment max ----------------
__global__ void k_attn(const float* __restrict__ q, const float* __restrict__ kk,
                       const float* __restrict__ ea, const long long* __restrict__ ei,
                       float* __restrict__ alpha, unsigned* __restrict__ amax,
                       int C, int HC) {
  int idx = blockIdx.x * blockDim.x + threadIdx.x;
  if (idx >= NEDGES * NHEADS) return;
  int e = idx >> 3, h = idx & 7;
  int s = (int)ei[e], d = (int)ei[NEDGES + e];
  const float* qp = q  + (size_t)d * HC + h * C;
  const float* kp = kk + (size_t)s * HC + h * C;
  const float* ep = ea + (size_t)e * HC + h * C;
  float acc = 0.f;
  for (int c = 0; c < C; c += 4) {
    float4 qv = *(const float4*)(qp + c);
    float4 kv = *(const float4*)(kp + c);
    float4 ev = *(const float4*)(ep + c);
    acc += qv.x * (kv.x + ev.x) + qv.y * (kv.y + ev.y) +
           qv.z * (kv.z + ev.z) + qv.w * (kv.w + ev.w);
  }
  acc *= rsqrtf((float)C);
  alpha[idx] = acc;
  atomicMax(&amax[d * NHEADS + h], flipf(acc));   // global_atomic_max_u32
}

// ---------------- softmax numerator + segment sum ----------------
__global__ void k_softmax_num(const long long* __restrict__ ei, float* __restrict__ alpha,
                              const unsigned* __restrict__ amax, float* __restrict__ denom) {
  int idx = blockIdx.x * blockDim.x + threadIdx.x;
  if (idx >= NEDGES * NHEADS) return;
  int e = idx >> 3, h = idx & 7;
  int d = (int)ei[NEDGES + e];
  float m = unflipf(amax[d * NHEADS + h]);
  float a = __expf(alpha[idx] - m);
  alpha[idx] = a;
  atomicAddF32(&denom[d * NHEADS + h], a);
}

// ---------------- weighted message scatter ----------------
__global__ void k_scatter(const float* __restrict__ v, const float* __restrict__ ea,
                          const float* __restrict__ alpha, const float* __restrict__ denom,
                          const long long* __restrict__ ei, float* __restrict__ acc,
                          int C, int HC) {
  long long idx = (long long)blockIdx.x * blockDim.x + threadIdx.x;
  if (idx >= (long long)NEDGES * HC) return;
  int e = (int)(idx / HC);
  int j = (int)(idx - (long long)e * HC);   // j = h*C + c
  int h = j / C;
  int s = (int)ei[e], d = (int)ei[NEDGES + e];
  float w = alpha[e * NHEADS + h] / (denom[d * NHEADS + h] + 1e-16f);
  float m = (v[(size_t)s * HC + j] + ea[(size_t)e * HC + j]) * w;
  atomicAddF32(&acc[(size_t)d * HC + j], m);   // global_atomic_add_f32
}

// ---------------- head mean + skip + LeakyReLU ----------------
__global__ void k_finalize(const float* __restrict__ acc, const float* __restrict__ skip,
                           float* __restrict__ xo, int C, int HC) {
  int idx = blockIdx.x * blockDim.x + threadIdx.x;
  if (idx >= NNODES * C) return;
  int n = idx / C, c = idx - n * C;
  float s = 0.f;
#pragma unroll
  for (int h = 0; h < NHEADS; ++h) s += acc[(size_t)n * HC + h * C + c];
  float o = s * (1.0f / NHEADS) + skip[idx];
  xo[idx] = (o > 0.f) ? o : NEG_SLP * o;
}

// ---------------- host orchestration ----------------
extern "C" void kernel_launch(void* const* d_in, const int* in_sizes, int n_in,
                              void* d_out, int out_size, void* d_ws, size_t ws_size,
                              hipStream_t stream) {
  (void)in_sizes; (void)n_in; (void)out_size; (void)ws_size;

  // inputs in setup_inputs() dict (insertion) order, params nested in literal order
  const float*     residue_atom = (const float*)d_in[0];
  const long long* edge_index   = (const long long*)d_in[1];  // int64 [2,E]
  const float*     edge_length  = (const float*)d_in[2];
  const float*     node_W       = (const float*)d_in[3];
  const float*     node_b       = (const float*)d_in[4];

  char* ws = (char*)d_ws;
  size_t off = 0;
  auto alloc = [&](size_t bytes) -> void* {
    void* p = ws + off;
    off += (bytes + 255) & ~(size_t)255;
    return p;
  };

  float* xa     = (float*)alloc((size_t)NNODES * MAXD * 4);
  float* xb     = (float*)alloc((size_t)NNODES * MAXD * 4);
  float* ea64   = (float*)alloc((size_t)NEDGES * EDGE_DIM * 4);
  float* qb     = (float*)alloc((size_t)NNODES * MAXHC * 4);
  float* kb     = (float*)alloc((size_t)NNODES * MAXHC * 4);
  float* vb     = (float*)alloc((size_t)NNODES * MAXHC * 4);
  float* skip   = (float*)alloc((size_t)NNODES * MAXD * 4);
  float* accb   = (float*)alloc((size_t)NNODES * MAXHC * 4);
  float* alpha  = (float*)alloc((size_t)NEDGES * NHEADS * 4);
  unsigned* amx = (unsigned*)alloc((size_t)NNODES * NHEADS * 4);
  float* denom  = (float*)alloc((size_t)NNODES * NHEADS * 4);
  float* ehc    = (float*)alloc((size_t)NEDGES * MAXHC * 4);
  __hip_bfloat16* wq_s = (__hip_bfloat16*)alloc((size_t)MAXD * MAXHC * 2);
  __hip_bfloat16* wk_s = (__hip_bfloat16*)alloc((size_t)MAXD * MAXHC * 2);
  __hip_bfloat16* wv_s = (__hip_bfloat16*)alloc((size_t)MAXD * MAXHC * 2);
  __hip_bfloat16* we_s = (__hip_bfloat16*)alloc((size_t)EDGE_DIM * MAXHC * 2);
  __hip_bfloat16* ws_s = (__hip_bfloat16*)alloc((size_t)MAXD * MAXD * 2);
  __hip_bfloat16* x_s  = (__hip_bfloat16*)alloc((size_t)NNODES * MAXD * 2);      // swizzled activations
  __hip_bfloat16* ea_s = (__hip_bfloat16*)alloc((size_t)NEDGES * EDGE_DIM * 2);  // swizzled edge attrs

  k_node_embed<<<(NNODES * RES_EMB + 255) / 256, 256, 0, stream>>>(residue_atom, node_W, node_b, xa);
  k_smear<<<(NEDGES * EDGE_DIM + 255) / 256, 256, 0, stream>>>(edge_length, ea64);
  // edge-attr A fragments are layer-invariant: swizzle once
  k_swizzle_a<<<(NEDGES * EDGE_DIM + 255) / 256, 256, 0, stream>>>(ea64, ea_s, NEDGES, EDGE_DIM);

  auto launch_gemm = [&](const __hip_bfloat16* A, const __hip_bfloat16* B, const float* bias,
                         float* Y, int M, int K, int Nc) {
    int groups = (M / 16) * (Nc / (16 * NT));
    k_wmma_gemm<<<(groups + 7) / 8, 256, 0, stream>>>(A, B, bias, Y, M, K, Nc);
  };

  const int dims[4] = {64, 128, 256, 256};
  float* xin = xa;
  float* xout = xb;
  for (int li = 0; li < 3; ++li) {
    const int dIn = dims[li], C = dims[li + 1], HC = NHEADS * C;
    const int base = 5 + li * 9;   // Wq,bq,Wk,bk,Wv,bv,We,Ws,bs
    const float* Wq  = (const float*)d_in[base + 0];
    const float* bq  = (const float*)d_in[base + 1];
    const float* Wk  = (const float*)d_in[base + 2];
    const float* bk_ = (const float*)d_in[base + 3];
    const float* Wv  = (const float*)d_in[base + 4];
    const float* bv  = (const float*)d_in[base + 5];
    const float* We  = (const float*)d_in[base + 6];
    const float* Wsk = (const float*)d_in[base + 7];
    const float* bs  = (const float*)d_in[base + 8];

    k_swizzle_w<<<(dIn * HC + 255) / 256, 256, 0, stream>>>(Wq, wq_s, dIn, HC);
    k_swizzle_w<<<(dIn * HC + 255) / 256, 256, 0, stream>>>(Wk, wk_s, dIn, HC);
    k_swizzle_w<<<(dIn * HC + 255) / 256, 256, 0, stream>>>(Wv, wv_s, dIn, HC);
    k_swizzle_w<<<(EDGE_DIM * HC + 255) / 256, 256, 0, stream>>>(We, we_s, EDGE_DIM, HC);
    k_swizzle_w<<<(dIn * C + 255) / 256, 256, 0, stream>>>(Wsk, ws_s, dIn, C);
    // swizzle this layer's input activations once; reused by q/k/v/skip GEMMs
    k_swizzle_a<<<(NNODES * dIn + 255) / 256, 256, 0, stream>>>(xin, x_s, NNODES, dIn);

    launch_gemm(x_s,  wq_s, bq,      qb,   NNODES, dIn,      HC);
    launch_gemm(x_s,  wk_s, bk_,     kb,   NNODES, dIn,      HC);
    launch_gemm(x_s,  wv_s, bv,      vb,   NNODES, dIn,      HC);
    launch_gemm(ea_s, we_s, nullptr, ehc,  NEDGES, EDGE_DIM, HC);
    launch_gemm(x_s,  ws_s, bs,      skip, NNODES, dIn,      C);

    hipMemsetAsync(amx,   0, (size_t)NNODES * NHEADS * 4, stream);
    hipMemsetAsync(denom, 0, (size_t)NNODES * NHEADS * 4, stream);
    hipMemsetAsync(accb,  0, (size_t)NNODES * HC * 4, stream);

    k_attn<<<(NEDGES * NHEADS + 255) / 256, 256, 0, stream>>>(qb, kb, ehc, edge_index,
                                                              alpha, amx, C, HC);
    k_softmax_num<<<(NEDGES * NHEADS + 255) / 256, 256, 0, stream>>>(edge_index, alpha, amx, denom);

    long long tot = (long long)NEDGES * HC;
    k_scatter<<<(unsigned)((tot + 255) / 256), 256, 0, stream>>>(vb, ehc, alpha, denom,
                                                                 edge_index, accb, C, HC);

    float* dst = (li == 2) ? (float*)d_out : xout;
    k_finalize<<<(NNODES * C + 255) / 256, 256, 0, stream>>>(accb, skip, dst, C, HC);

    float* t = xin; xin = xout; xout = t;
  }
}
